// GravConv_15719580303533
// MI455X (gfx1250) — matrix operand order repaired and checked by reading, not compiled, not run on hardware
//
#include <hip/hip_runtime.h>
#include <hip/hip_bf16.h>
#include <float.h>

typedef __attribute__((ext_vector_type(16))) _Float16 v16h;
typedef __attribute__((ext_vector_type(8)))  _Float16 v8h;
typedef __attribute__((ext_vector_type(8)))  float    v8f;

#define NN   50000      // N_NODES
#define NE   800000     // N_EDGES
#define HID  128        // EMB_HIDDEN
#define H1   129        // HID+1
#define H1P  132        // fp32 padded row stride for hidden/agg
#define K0P  160        // fp16 padded K for first spatial layer (129 -> 160)
#define EMB  12         // EMB_DIM
#define CATP 544        // fp16 padded K for first feature layer (516 -> 544)
#define RINV (1.0f/0.09f)   // grav_w / r^2 = 1/0.3^2

// ---------------------------------------------------------------- utilities

__device__ __forceinline__ float atomicMaxF(float* addr, float val) {
  // sign-split trick: lowers to global_atomic_max_i32 / min_u32 resolved in L2
  if (val >= 0.0f)
    return __int_as_float(atomicMax((int*)addr, __float_as_int(val)));
  else
    return __uint_as_float(atomicMin((unsigned int*)addr, __float_as_uint(val)));
}

// low 32 bits of a generic pointer to __shared__ = workgroup-relative LDS addr
__device__ __forceinline__ unsigned ldsAddrOf(const void* p) {
  return (unsigned)(uintptr_t)p;
}

// async copy 16B from global to LDS (per-lane), tracked by ASYNCcnt
__device__ __forceinline__ void asyncCopy16(unsigned ldsAddr, const void* gptr) {
  asm volatile("global_load_async_to_lds_b128 %0, %1, off"
               :: "v"(ldsAddr), "v"(gptr) : "memory");
}
__device__ __forceinline__ void asyncWaitAll() {
  asm volatile("s_wait_asynccnt 0" ::: "memory");
}

// ------------------------------------------------- 1. hidden = [x, mean(x)]

__global__ __launch_bounds__(128) void k_hidden(
    const float* __restrict__ x, float* __restrict__ hf32,
    _Float16* __restrict__ hf16)
{
  __shared__ float wsum[4];
  __shared__ float smean;
  const int row = blockIdx.x;
  const int t = threadIdx.x;
  float v = x[(size_t)row * HID + t];
  float s = v;
  for (int o = 16; o > 0; o >>= 1) s += __shfl_down(s, o, 32);
  if ((t & 31) == 0) wsum[t >> 5] = s;
  __syncthreads();
  if (t == 0) smean = (wsum[0] + wsum[1] + wsum[2] + wsum[3]) * (1.0f / HID);
  __syncthreads();
  const float mean = smean;
  hf32[(size_t)row * H1P + t] = v;
  hf16[(size_t)row * K0P + t] = (_Float16)v;
  if (t < 32) {                       // cols 128..159 : mean then zero-pad
    float pv = (t == 0) ? mean : 0.0f;
    hf16[(size_t)row * K0P + HID + t] = (_Float16)pv;
    if (t < 4) hf32[(size_t)row * H1P + HID + t] = pv;
  }
}

// ---------------------------------------- 2. weight repack fp32 -> fp16 pad

__global__ void k_packW(const float* __restrict__ W, int K, int Ncols,
                        _Float16* __restrict__ out, int Kpad, int Npad)
{
  const int total = Kpad * Npad;
  for (int i = blockIdx.x * blockDim.x + threadIdx.x; i < total;
       i += gridDim.x * blockDim.x) {
    int k = i / Npad, n = i % Npad;
    float v = (k < K && n < Ncols) ? W[(size_t)k * Ncols + n] : 0.0f;
    out[i] = (_Float16)v;
  }
}

// ------------------------- 3. WMMA GEMM, 16 rows x 128 cols per block.
// A tile (16 x Kpad fp16) is staged ONCE into LDS with async global->LDS
// copies (ASYNCcnt), removing the 8x per-wave redundancy; WMMA A-fragments
// are then served by ds_load_b128. B (weights) streams from global/L2 with
// prefetch. Two independent accumulators break the WMMA RAW chain.
// Epilogue: bias + LayerNorm + SiLU fused in LDS.

__global__ __launch_bounds__(256) void k_gemm128(
    const _Float16* __restrict__ A, int Kpad,
    const _Float16* __restrict__ W,
    const float* __restrict__ bias,
    const float* __restrict__ gamma,
    const float* __restrict__ beta,
    _Float16* __restrict__ out16,
    float* __restrict__ out32,
    int ln_silu)
{
  __shared__ _Float16 shA[16 * CATP];          // max 17408 B
  __shared__ float tile[16][HID + 4];
  __shared__ float psum[16][16];
  __shared__ float psum2[16][16];
  __shared__ float rmean[16], rrstd[16];

  const int m0   = blockIdx.x * 16;
  const int wave = threadIdx.x >> 5;
  const int lane = threadIdx.x & 31;
  const int n0   = wave * 16;

  // ---- cooperative async stage of the 16 x Kpad A tile into LDS ----
  const int chunksPerRow = Kpad >> 3;          // 16B chunks per row
  const int totChunks = 16 * chunksPerRow;
  const unsigned shABase = ldsAddrOf(shA);
  for (int c = threadIdx.x; c < totChunks; c += 256) {
    const int r  = c / chunksPerRow;
    const int co = c - r * chunksPerRow;       // chunk within row
    const _Float16* gp = A + (size_t)(m0 + r) * Kpad + co * 8;
    asyncCopy16(shABase + (unsigned)((r * Kpad + co * 8) * 2), gp);
  }
  asyncWaitAll();
  __syncthreads();

  // ---- WMMA K loop (A from LDS, B from global) ----
  // A fragment per ISA 7.12.2 (16-bit A 16x32):
  // lanes 0-15: row=lane, K halfs {kb..kb+7, kb+16..kb+23};
  // lanes 16-31: row=lane-16, K halfs {kb+8..kb+15, kb+24..kb+31}
  const int ar = lane & 15;
  const int aK = (lane >> 4) * 8;
  const _Float16* aRowP = shA + ar * Kpad;

  v8f acc0 = {}, acc1 = {};
  int kb = 0;
  for (; kb + 64 <= Kpad; kb += 64) {
    union { v16h v; v8h h[2]; } a0, a1;
    a0.h[0] = *(const v8h*)(aRowP + kb + aK);
    a0.h[1] = *(const v8h*)(aRowP + kb + aK + 16);
    a1.h[0] = *(const v8h*)(aRowP + kb + 32 + aK);
    a1.h[1] = *(const v8h*)(aRowP + kb + 32 + aK + 16);
    v16h b0, b1;
    for (int s = 0; s < 16; ++s) {
      const int kk = aK + (s & 7) + (s >> 3) * 16;
      b0[s] = W[(size_t)(kb + kk) * HID + n0 + (lane & 15)];
      b1[s] = W[(size_t)(kb + 32 + kk) * HID + n0 + (lane & 15)];
    }
    if (kb + 64 < Kpad)
      __builtin_prefetch(W + (size_t)(kb + 64 + lane) * HID + n0, 0, 3);
    acc0 = __builtin_amdgcn_wmma_f32_16x16x32_f16(false, a0.v, false, b0,
                                                  (short)0, acc0, false, false);
    acc1 = __builtin_amdgcn_wmma_f32_16x16x32_f16(false, a1.v, false, b1,
                                                  (short)0, acc1, false, false);
  }
  if (kb < Kpad) {                              // K tail (Kpad % 64 == 32)
    union { v16h v; v8h h[2]; } a0;
    a0.h[0] = *(const v8h*)(aRowP + kb + aK);
    a0.h[1] = *(const v8h*)(aRowP + kb + aK + 16);
    v16h b0;
    for (int s = 0; s < 16; ++s)
      b0[s] = W[(size_t)(kb + aK + (s & 7) + (s >> 3) * 16) * HID + n0 + (lane & 15)];
    acc0 = __builtin_amdgcn_wmma_f32_16x16x32_f16(false, a0.v, false, b0,
                                                  (short)0, acc0, false, false);
  }
#pragma unroll
  for (int v = 0; v < 8; ++v) acc0[v] += acc1[v];

  // C layout: vgpr v, lanes 0-15 -> M=v, N=lane; lanes 16-31 -> M=v+8
  const float bv = bias[n0 + (lane & 15)];
  const int rl = (lane >> 4) * 8;
  const int cl = n0 + (lane & 15);
#pragma unroll
  for (int v = 0; v < 8; ++v) tile[rl + v][cl] = acc0[v] + bv;
  __syncthreads();

  if (ln_silu) {
    const int r = threadIdx.x >> 4;
    const int c16 = threadIdx.x & 15;
    float s = 0.f, s2 = 0.f;
#pragma unroll
    for (int c = c16; c < HID; c += 16) {
      float y = tile[r][c]; s += y; s2 += y * y;
    }
    psum[r][c16] = s; psum2[r][c16] = s2;
    __syncthreads();
    if (threadIdx.x < 16) {
      float ts = 0.f, ts2 = 0.f;
      for (int j = 0; j < 16; ++j) { ts += psum[threadIdx.x][j]; ts2 += psum2[threadIdx.x][j]; }
      float mu  = ts * (1.0f / HID);
      float var = ts2 * (1.0f / HID) - mu * mu;
      rmean[threadIdx.x] = mu;
      rrstd[threadIdx.x] = rsqrtf(var + 1e-5f);
    }
    __syncthreads();
    for (int i = threadIdx.x; i < 16 * HID; i += 256) {
      int rr = i >> 7, cc = i & 127;
      float y = (tile[rr][cc] - rmean[rr]) * rrstd[rr] * gamma[cc] + beta[cc];
      y = y / (1.0f + __expf(-y));       // SiLU
      out16[(size_t)(m0 + rr) * HID + cc] = (_Float16)y;
    }
  } else {
    for (int i = threadIdx.x; i < 16 * HID; i += 256) {
      int rr = i >> 7, cc = i & 127;
      out32[(size_t)(m0 + rr) * HID + cc] = tile[rr][cc];
    }
  }
}

// ----------------- 4. last spatial layer 128 -> 12 (N padded to 16) + L2 norm

__global__ __launch_bounds__(32) void k_spatial_out(
    const _Float16* __restrict__ A,   // [NN,128] fp16
    const _Float16* __restrict__ W,   // [128,16] fp16, cols 12..15 zero
    const float* __restrict__ bias,   // [12]
    float* __restrict__ spatial)      // [NN,12] fp32 (d_out region 2)
{
  __shared__ float tile[16][17];
  const int m0 = blockIdx.x * 16;
  const int lane = threadIdx.x & 31;
  const int aRow = m0 + (lane & 15);
  const int aK = (lane >> 4) * 8;
  const size_t aBase = (size_t)aRow * HID;
  v8f acc = {};
  for (int kb = 0; kb < HID; kb += 32) {
    union { v16h v; v8h h[2]; } a;
    a.h[0] = *(const v8h*)(A + aBase + kb + aK);
    a.h[1] = *(const v8h*)(A + aBase + kb + aK + 16);
    v16h b;
    for (int s = 0; s < 16; ++s)
      b[s] = W[(size_t)(kb + aK + (s & 7) + (s >> 3) * 16) * 16 + (lane & 15)];
    acc = __builtin_amdgcn_wmma_f32_16x16x32_f16(false, a.v, false, b,
                                                 (short)0, acc, false, false);
  }
  const int col = lane & 15;
  const float bv = (col < EMB) ? bias[col] : 0.0f;
  const int rl = (lane >> 4) * 8;
#pragma unroll
  for (int v = 0; v < 8; ++v) tile[rl + v][col] = acc[v] + bv;
  __syncthreads();                     // single-wave block: cheap
  if (lane < 16) {
    float ss = 0.f;
#pragma unroll
    for (int j = 0; j < EMB; ++j) { float y = tile[lane][j]; ss += y * y; }
    float inv = 1.0f / fmaxf(sqrtf(ss), 1e-12f);
#pragma unroll
    for (int j = 0; j < EMB; ++j)
      spatial[(size_t)(m0 + lane) * EMB + j] = tile[lane][j] * inv;
  }
}

// ------------------------------------------------ 5. aggregation scratch init

__global__ void k_zero_agg(float* __restrict__ aggSum, float* __restrict__ aggMax,
                           unsigned int* __restrict__ counts)
{
  const int total = NN * H1P;
  for (int i = blockIdx.x * blockDim.x + threadIdx.x; i < total;
       i += gridDim.x * blockDim.x) {
    aggSum[i] = 0.0f;
    aggMax[i] = -3.402823466e38f;
    if (i < NN) counts[i] = 0u;
  }
}

// ---------------------------------- 6. edge scatter: one wave32 per edge

__global__ __launch_bounds__(256) void k_edges(
    const int* __restrict__ edges, const float* __restrict__ sp,
    const float* __restrict__ hf32, float* __restrict__ aggSum,
    float* __restrict__ aggMax, unsigned int* __restrict__ counts)
{
  const int lane = threadIdx.x & 31;
  const int gw = blockIdx.x * (blockDim.x >> 5) + (threadIdx.x >> 5);
  const int nw = gridDim.x * (blockDim.x >> 5);
  for (int e = gw; e < NE; e += nw) {
    const int s = edges[e];
    const int t = edges[NE + e];
    float d = 0.0f;
    if (lane < EMB) {
      float a = sp[(size_t)s * EMB + lane] - sp[(size_t)t * EMB + lane];
      d = a * a;
    }
    for (int o = 8; o > 0; o >>= 1) d += __shfl_down(d, o, 32);
    d = __shfl(d, 0, 32);
    const float w = __expf(-d * RINV);
    for (int c = lane; c < H1; c += 32) {
      float m = hf32[(size_t)s * H1P + c] * w;
      atomicAdd(&aggSum[(size_t)t * H1P + c], m);   // L2-resident atomics
      atomicMaxF(&aggMax[(size_t)t * H1P + c], m);
    }
    if (lane == 0) atomicAdd(&counts[t], 1u);
  }
}

// --------------- 7. finalize mean/max + build fp16 concat row [sum|mean|max|hid]

__global__ __launch_bounds__(256) void k_concat(
    const float* __restrict__ aggSum, const float* __restrict__ aggMax,
    const unsigned int* __restrict__ counts, const float* __restrict__ hf32,
    _Float16* __restrict__ cat16)
{
  const int row = blockIdx.x;
  const float cnt = (float)counts[row];
  const float inv = 1.0f / fmaxf(cnt, 1.0f);
  const bool has = cnt > 0.0f;
  for (int c = threadIdx.x; c < CATP; c += blockDim.x) {
    float v;
    if (c < H1)          v = aggSum[(size_t)row * H1P + c];
    else if (c < 2 * H1) v = aggSum[(size_t)row * H1P + (c - H1)] * inv;
    else if (c < 3 * H1) v = has ? aggMax[(size_t)row * H1P + (c - 2 * H1)] : 0.0f;
    else if (c < 4 * H1) v = hf32[(size_t)row * H1P + (c - 3 * H1)];
    else                 v = 0.0f;
    cat16[(size_t)row * CATP + c] = (_Float16)v;
  }
}

// ------------------------------- 8. secondary outputs: edge_index + grav_w

__global__ void k_copy_edges(const int* __restrict__ edges,
                             float* __restrict__ outE, float* __restrict__ outGrav)
{
  const int total = 2 * NE;
  int gid = blockIdx.x * blockDim.x + threadIdx.x;
  if (gid == 0) outGrav[0] = 1.0f;
  for (int i = gid; i < total; i += gridDim.x * blockDim.x)
    outE[i] = (float)edges[i];
}

// --------------------------------------------------------------------- host

extern "C" void kernel_launch(void* const* d_in, const int* in_sizes, int n_in,
                              void* d_out, int out_size, void* d_ws, size_t ws_size,
                              hipStream_t stream) {
  (void)in_sizes; (void)n_in; (void)out_size; (void)ws_size;
  // setup_inputs dict order:
  // 0 x, 1 edge_index, 2 current_epoch,
  // 3-6 spatial_Ws, 7-10 spatial_bs, 11-13 spatial_gs, 14-16 spatial_betas,
  // 17-19 feat_Ws, 20-22 feat_bs, 23-24 feat_gs, 25-26 feat_betas
  const float* x     = (const float*)d_in[0];
  const int*   edges = (const int*)d_in[1];
  const float* sW[4] = {(const float*)d_in[3], (const float*)d_in[4],
                        (const float*)d_in[5], (const float*)d_in[6]};
  const float* sB[4] = {(const float*)d_in[7], (const float*)d_in[8],
                        (const float*)d_in[9], (const float*)d_in[10]};
  const float* sG[3] = {(const float*)d_in[11], (const float*)d_in[12],
                        (const float*)d_in[13]};
  const float* sBe[3]= {(const float*)d_in[14], (const float*)d_in[15],
                        (const float*)d_in[16]};
  const float* fW[3] = {(const float*)d_in[17], (const float*)d_in[18],
                        (const float*)d_in[19]};
  const float* fB[3] = {(const float*)d_in[20], (const float*)d_in[21],
                        (const float*)d_in[22]};
  const float* fG[2] = {(const float*)d_in[23], (const float*)d_in[24]};
  const float* fBe[2]= {(const float*)d_in[25], (const float*)d_in[26]};

  // workspace partition (256B aligned slabs)
  char* wsb = (char*)d_ws;
  size_t off = 0;
  auto take = [&](size_t bytes) -> void* {
    void* p = wsb + off;
    off += (bytes + 255) & ~(size_t)255;
    return p;
  };
  float*        hf32   = (float*)take((size_t)NN * H1P * 4);
  _Float16*     hf16   = (_Float16*)take((size_t)NN * K0P * 2);
  _Float16*     actA   = (_Float16*)take((size_t)NN * HID * 2);
  _Float16*     actB   = (_Float16*)take((size_t)NN * HID * 2);
  _Float16*     cat16  = (_Float16*)take((size_t)NN * CATP * 2);
  float*        aggSum = (float*)take((size_t)NN * H1P * 4);
  float*        aggMax = (float*)take((size_t)NN * H1P * 4);
  unsigned int* counts = (unsigned int*)take((size_t)NN * 4);
  _Float16*     wsp0   = (_Float16*)take((size_t)K0P * HID * 2);
  _Float16*     wsp1   = (_Float16*)take((size_t)HID * HID * 2);
  _Float16*     wsp2   = (_Float16*)take((size_t)HID * HID * 2);
  _Float16*     wsp3   = (_Float16*)take((size_t)HID * 16 * 2);
  _Float16*     wf0    = (_Float16*)take((size_t)CATP * HID * 2);
  _Float16*     wf1    = (_Float16*)take((size_t)HID * HID * 2);
  _Float16*     wf2    = (_Float16*)take((size_t)HID * HID * 2);

  // d_out layout: [out 50000*128][edge_index 2*800000][spatial 50000*12][grav 1]
  float* out_feat    = (float*)d_out;
  float* out_edges   = out_feat + (size_t)NN * HID;
  float* out_spatial = out_edges + (size_t)2 * NE;
  float* out_grav    = out_spatial + (size_t)NN * EMB;

  const int GEMM_GRID = NN / 16;   // 3125, exact

  // 1) hidden = [x, mean(x)] in fp32 (stride 132) and fp16 (stride 160)
  k_hidden<<<NN, HID, 0, stream>>>(x, hf32, hf16);

  // 2) repack all weights to fp16, zero-padded
  k_packW<<<128, 256, 0, stream>>>(sW[0], H1,  HID, wsp0, K0P, HID);
  k_packW<<<128, 256, 0, stream>>>(sW[1], HID, HID, wsp1, HID, HID);
  k_packW<<<128, 256, 0, stream>>>(sW[2], HID, HID, wsp2, HID, HID);
  k_packW<<<16,  256, 0, stream>>>(sW[3], HID, EMB, wsp3, HID, 16);
  k_packW<<<256, 256, 0, stream>>>(fW[0], 4 * H1, HID, wf0, CATP, HID);
  k_packW<<<128, 256, 0, stream>>>(fW[1], HID, HID, wf1, HID, HID);
  k_packW<<<128, 256, 0, stream>>>(fW[2], HID, HID, wf2, HID, HID);

  // 3) spatial MLP: 3x (Linear+LN+SiLU) then 128->12 + L2 normalize
  k_gemm128<<<GEMM_GRID, 256, 0, stream>>>(hf16, K0P, wsp0, sB[0], sG[0], sBe[0],
                                           actA, nullptr, 1);
  k_gemm128<<<GEMM_GRID, 256, 0, stream>>>(actA, HID, wsp1, sB[1], sG[1], sBe[1],
                                           actB, nullptr, 1);
  k_gemm128<<<GEMM_GRID, 256, 0, stream>>>(actB, HID, wsp2, sB[2], sG[2], sBe[2],
                                           actA, nullptr, 1);
  k_spatial_out<<<GEMM_GRID, 32, 0, stream>>>(actA, wsp3, sB[3], out_spatial);

  // 4) edge scatter (sum / count / max) over L2-resident buffers
  k_zero_agg<<<4096, 256, 0, stream>>>(aggSum, aggMax, counts);
  k_edges<<<2048, 256, 0, stream>>>(edges, out_spatial, hf32, aggSum, aggMax, counts);

  // 5) secondary outputs
  k_copy_edges<<<4096, 256, 0, stream>>>(edges, out_edges, out_grav);

  // 6) finalize aggregation + build fp16 concat [sum|mean|max|hidden] (pad 544)
  k_concat<<<NN, 256, 0, stream>>>(aggSum, aggMax, counts, hf32, cat16);

  // 7) feature MLP: 2x (Linear+LN+SiLU) + bare Linear -> d_out region 0
  k_gemm128<<<GEMM_GRID, 256, 0, stream>>>(cat16, CATP, wf0, fB[0], fG[0], fBe[0],
                                           actA, nullptr, 1);
  k_gemm128<<<GEMM_GRID, 256, 0, stream>>>(actA, HID, wf1, fB[1], fG[1], fBe[1],
                                           actB, nullptr, 1);
  k_gemm128<<<GEMM_GRID, 256, 0, stream>>>(actB, HID, wf2, fB[2], nullptr, nullptr,
                                           nullptr, out_feat, 0);
}
